// Model_79680233275459
// MI455X (gfx1250) — matrix operand order repaired and checked by reading, not compiled, not run on hardware
//
#include <hip/hip_runtime.h>
#include <hip/hip_bf16.h>

// ---------------------------------------------------------------------------
// DGCNN forward for MI455X (gfx1250, wave32).  All contractions use
// V_WMMA_F32_16X16X32_BF16.  All WMMA operands live in point-major bf16
// buffers so each lane's fragment is two 16B vector loads (global_load_b128),
// eliminating the v_mov_b16 packing VALU.
// Everything stays L2-resident (192MB L2); BN is two-pass with GEMM recompute.
// ---------------------------------------------------------------------------

typedef __attribute__((ext_vector_type(16))) __bf16 v16bf;
typedef __attribute__((ext_vector_type(8)))  float  v8f;

#define KNN_K 20

// ---- bf16 helpers ---------------------------------------------------------
__device__ __forceinline__ unsigned short f2bf(float f) {
  unsigned u = __builtin_bit_cast(unsigned, f);
  unsigned r = (u + 0x7FFFu + ((u >> 16) & 1u)) >> 16;   // round-to-nearest-even
  return (unsigned short)r;
}
__device__ __forceinline__ __bf16 bfbits(unsigned short s) {
  return __builtin_bit_cast(__bf16, s);
}

union Frag { v16bf v; uint4 u[2]; };

// A-operand fragment (contiguous k order, matched by B): lane holds 16 contiguous halves
// {c..c+15} of its row, where c = kb + ((lane&16)?16:0).
__device__ __forceinline__ v16bf load_afrag(const unsigned short* p) {
  Frag t;
  t.u[0] = *(const uint4*)(p);
  t.u[1] = *(const uint4*)(p + 8);
  return t.v;
}
// B-operand fragment (contiguous k order, matched by A): lane holds 16 contiguous halves
// starting at c = kb + ((lane&16)?16:0) of its column.
__device__ __forceinline__ v16bf load_bfrag(const unsigned short* p) {
  Frag t;
  t.u[0] = *(const uint4*)(p);
  t.u[1] = *(const uint4*)(p + 8);
  return t.v;
}

// ---- order-preserving float<->uint encoding for atomicMax max-pool --------
__device__ __forceinline__ unsigned enc_f32(float v) {
  unsigned u = __builtin_bit_cast(unsigned, v);
  return (u & 0x80000000u) ? ~u : (u | 0x80000000u);
}
__device__ __forceinline__ float dec_f32(unsigned e) {
  unsigned u = (e & 0x80000000u) ? (e & 0x7FFFFFFFu) : ~e;
  return __builtin_bit_cast(float, u);
}

// ===========================================================================
// prep_input: x [B,C,N] -> point-major bf16 F [B,N,64] (zero padded) + norms.
// ===========================================================================
__global__ void prep_input(const float* __restrict__ x, unsigned short* __restrict__ F,
                           float* __restrict__ sq, int B, int N, int C) {
  int t = blockIdx.x * blockDim.x + threadIdx.x;
  if (t >= B * N) return;
  int b = t / N, n = t - b * N;
  float acc = 0.f;
  unsigned short* dst = F + (size_t)t * 64;
  for (int c = 0; c < 64; ++c) {
    float f = (c < C) ? x[((size_t)(b * C + c)) * N + n] : 0.0f;
    dst[c] = f2bf(f);
    acc += f * f;
  }
  sq[t] = acc;
}

// ===========================================================================
// cvt_weights: fp32 W [O,Cin] -> bf16 Wb [O,Keff] zero-padded.
// ===========================================================================
__global__ void cvt_weights(const float* __restrict__ W, unsigned short* __restrict__ Wb,
                            int O, int Cin, int Keff) {
  int t = blockIdx.x * blockDim.x + threadIdx.x;
  if (t >= O * Keff) return;
  int o = t / Keff, k = t - o * Keff;
  Wb[t] = (k < Cin) ? f2bf(W[o * Cin + k]) : (unsigned short)0;
}

// ===========================================================================
// knn_kernel: per block = 16 query points.  WMMA computes 16x256 neg-distance
// strips into LDS; 256 threads do segmented top-20 + 16-way merge -> idx.
// F is point-major bf16 [B,N,64]: all fragments are b128 vector loads.
// ===========================================================================
__global__ __launch_bounds__(256) void knn_kernel(const unsigned short* __restrict__ F,
                                                  const float* __restrict__ sq,
                                                  int* __restrict__ nbr, int N) {
  __shared__ float s_dist[16 * 256];                 // 16 KB strip
  __shared__ float s_cval[16][16][KNN_K];            // 20 KB candidates
  __shared__ int   s_cidx[16][16][KNN_K];            // 20 KB candidates
  const int b    = blockIdx.y;
  const int i0   = blockIdx.x * 16;
  const int tid  = threadIdx.x;
  const int lane = tid & 31;
  const int wave = tid >> 5;
  const int m    = lane & 15;
  const int coA  = (lane & 16) ? 16 : 0;
  const int coB  = (lane & 16) ? 16 : 0;

  // Per-wave A operand (16 query points x 64 channels), block-invariant.
  const unsigned short* arow = F + ((size_t)(b * N) + i0 + m) * 64;
  v16bf a0 = load_afrag(arow + coA);        // K  0..31
  v16bf a1 = load_afrag(arow + 32 + coA);   // K 32..63

  const int row = tid >> 4;   // 0..15: query row this thread scans
  const int cls = tid & 15;   // column class (mod 16)
  float tv[KNN_K];
  int   ti[KNN_K];
#pragma unroll
  for (int q = 0; q < KNN_K; ++q) { tv[q] = -3.0e38f; ti[q] = 0; }

  for (int chunk = 0; chunk < N; chunk += 256) {
    if (chunk + 256 < N)  // hint next chunk of F into cache (global_prefetch)
      __builtin_prefetch(&F[((size_t)(b * N) + chunk + 256) * 64 + lane * 4], 0, 0);
    // --- compute phase: each wave fills 2 of the 16 column tiles ------------
    for (int t2 = wave * 2; t2 < wave * 2 + 2; ++t2) {
      const int j0  = chunk + t2 * 16;
      const int col = lane & 15;
      const unsigned short* bcol = F + ((size_t)(b * N) + j0 + col) * 64;
      v16bf b0 = load_bfrag(bcol + coB);        // K  0..31
      v16bf b1 = load_bfrag(bcol + 32 + coB);   // K 32..63
      v8f c = {};
      c = __builtin_amdgcn_wmma_f32_16x16x32_bf16(false, a0, false, b0, (short)0, c, false, false);
      c = __builtin_amdgcn_wmma_f32_16x16x32_bf16(false, a1, false, b1, (short)0, c, false, false);
      const float sqj = sq[b * N + j0 + col];
#pragma unroll
      for (int i = 0; i < 8; ++i) {
        int mr = i + ((lane & 16) ? 8 : 0);
        s_dist[mr * 256 + t2 * 16 + col] = 2.0f * c[i] - sq[b * N + i0 + mr] - sqj;
      }
    }
    __syncthreads();
    // --- scan phase: all 256 threads maintain per-segment top-20 -----------
    for (int w2 = 0; w2 < 16; ++w2) {
      int jl = cls + 16 * w2;
      float v = s_dist[row * 256 + jl];
      if (v > tv[KNN_K - 1]) {
        int pos = KNN_K - 1;
        while (pos > 0 && tv[pos - 1] < v) { tv[pos] = tv[pos - 1]; ti[pos] = ti[pos - 1]; --pos; }
        tv[pos] = v; ti[pos] = chunk + jl;
      }
    }
    __syncthreads();
  }

  // --- merge: 16 sorted lists per row -> global top-20 ----------------------
#pragma unroll
  for (int q = 0; q < KNN_K; ++q) { s_cval[row][cls][q] = tv[q]; s_cidx[row][cls][q] = ti[q]; }
  __syncthreads();
  if (tid < 16) {
    int heads[16];
#pragma unroll
    for (int s = 0; s < 16; ++s) heads[s] = 0;
    for (int q = 0; q < KNN_K; ++q) {
      float best = -3.4e38f; int bs = 0;
      for (int s = 0; s < 16; ++s) {
        int h = heads[s];
        if (h < KNN_K) { float v = s_cval[tid][s][h]; if (v > best) { best = v; bs = s; } }
      }
      nbr[((size_t)(b * N + i0 + tid)) * KNN_K + q] = s_cidx[tid][bs][heads[bs]];
      heads[bs]++;
    }
  }
}

// ===========================================================================
// edge_gemm<MODE,PASS>: one wave per 16x16 output tile of [O x (N*20)].
//   MODE 0: block-0 edge features [x_i ; x_j - x_i], Cin=6 (scalar gather).
//   MODE 1: edge features from point-major bf16 map Fs [B,N,64], Cin=64:
//           K blocks 0..63 are x_i (direct b128 frags), 64..127 are x_j-x_i.
//   MODE 2: dense bf16 intermediate t [B,N*20,64] (direct b128 frags).
//   PASS 1: accumulate BN batch stats;  PASS 2: BN+LeakyReLU, then store
//           bf16 t (vector store) and/or max-pool via encoded atomicMax.
// ===========================================================================
template <int MODE, int PASS>
__global__ __launch_bounds__(256) void edge_gemm(
    const unsigned short* __restrict__ Wb,
    const float* __restrict__ X, const int* __restrict__ nbr,
    const unsigned short* __restrict__ Fs, const unsigned short* __restrict__ T,
    float* __restrict__ stats, const float* __restrict__ scsh,
    unsigned short* __restrict__ tOut, unsigned int* __restrict__ encOut,
    int B, int N, int O, int Keff) {
  const int lane = threadIdx.x & 31;
  const int wid  = (blockIdx.x * blockDim.x + threadIdx.x) >> 5;
  const int NK = N * KNN_K;
  const int coltiles = NK >> 4;
  const int perB = coltiles * (O >> 4);
  const int b = wid / perB;
  if (b >= B) return;                       // wave-uniform: EXEC stays all-ones
  const int rem  = wid - b * perB;
  const int row0 = (rem / coltiles) << 4;
  const int col0 = (rem % coltiles) << 4;

  // Per-lane column info (column fixed per lane within the tile).
  const int colg = col0 + (lane & 15);
  const int n  = colg / KNN_K;
  const int kk = colg - n * KNN_K;
  int j = 0;
  if (MODE == 0 || MODE == 1) j = nbr[((size_t)(b * N + n)) * KNN_K + kk];
  const int m   = lane & 15;
  const int coA = (lane & 16) ? 16 : 0;
  const int coB = (lane & 16) ? 16 : 0;
  const unsigned short* wrow = Wb + (size_t)(row0 + m) * Keff;

  v8f c = {};
  for (int kb = 0; kb < Keff; kb += 32) {
    v16bf a = load_afrag(wrow + kb + coA);
    v16bf bb;
    if (MODE == 0) {
      // Cin=6: p<6 -> x_i, p<12 -> x_j - x_i, else 0 (Keff==32, one step).
#pragma unroll
      for (int h = 0; h < 16; ++h) {
        int p = coB + h;
        float xf = 0.0f;
        if (p < 6)       xf = X[((size_t)(b * 6 + p)) * N + n];
        else if (p < 12) { int cc = p - 6;
                           xf = X[((size_t)(b * 6 + cc)) * N + j] -
                                X[((size_t)(b * 6 + cc)) * N + n]; }
        bb[h] = bfbits(f2bf(xf));
      }
    } else if (MODE == 1) {
      const int cc = kb + coB;              // start channel of this 16-run
      const unsigned short* pn = Fs + ((size_t)(b * N) + n) * 64;
      if (cc < 64) {                        // x_i part: direct fragment
        bb = load_bfrag(pn + cc);
      } else {                              // (x_j - x_i) part
        const unsigned short* pj = Fs + ((size_t)(b * N) + j) * 64;
        v16bf xi = load_bfrag(pn + (cc - 64));
        v16bf xj = load_bfrag(pj + (cc - 64));
#pragma unroll
        for (int h = 0; h < 16; ++h)
          bb[h] = (__bf16)((float)xj[h] - (float)xi[h]);
      }
    } else {                                // MODE 2: dense t
      bb = load_bfrag(T + ((size_t)(b * NK) + colg) * 64 + kb + coB);
    }
    c = __builtin_amdgcn_wmma_f32_16x16x32_bf16(false, a, false, bb, (short)0, c, false, false);
  }

  if (PASS == 1) {
#pragma unroll
    for (int i = 0; i < 8; ++i) {
      float vs = c[i], vq = c[i] * c[i];
#pragma unroll
      for (int msk = 1; msk < 16; msk <<= 1) { vs += __shfl_xor(vs, msk); vq += __shfl_xor(vq, msk); }
      if ((lane & 15) == 0) {
        int rowg = row0 + i + ((lane & 16) ? 8 : 0);
        atomicAdd(&stats[rowg], vs);
        atomicAdd(&stats[O + rowg], vq);
      }
    }
  } else {
    const int rbase = row0 + ((lane & 16) ? 8 : 0);
    if (tOut) {
      union { uint4 q; unsigned short s[8]; } pk;
#pragma unroll
      for (int i = 0; i < 8; ++i) {
        int rowg = rbase + i;
        float v = c[i] * scsh[rowg] + scsh[O + rowg];
        v = (v >= 0.0f) ? v : 0.2f * v;
        pk.s[i] = f2bf(v);
      }
      // point-major t: lane's 8 channels are contiguous -> one b128 store
      *(uint4*)(tOut + ((size_t)(b * NK) + colg) * 64 + rbase) = pk.q;
    }
    if (encOut) {
#pragma unroll
      for (int i = 0; i < 8; ++i) {
        int rowg = rbase + i;
        float v = c[i] * scsh[rowg] + scsh[O + rowg];
        v = (v >= 0.0f) ? v : 0.2f * v;
        atomicMax(&encOut[((size_t)(b * 64 + rowg)) * N + n], enc_f32(v));
      }
    }
  }
}

// ===========================================================================
// bn_finalize: sums -> (scale, shift) per channel.
// ===========================================================================
__global__ void bn_finalize(const float* __restrict__ stats, const float* __restrict__ g,
                            const float* __restrict__ bb, float* __restrict__ scsh,
                            float invcnt, int O) {
  int o = threadIdx.x;
  if (o < O) {
    float mean = stats[o] * invcnt;
    float var  = stats[O + o] * invcnt - mean * mean;
    float sc   = g[o] * rsqrtf(var + 1e-5f);
    scsh[o]     = sc;
    scsh[O + o] = bb[o] - mean * sc;
  }
}

// ===========================================================================
// decode_max: decode uint-encoded max-pool [B,64,N] -> point-major bf16 map
// xb [B,N,64] + squared norms for the next KNN.
// ===========================================================================
__global__ void decode_max(const unsigned int* __restrict__ enc, unsigned short* __restrict__ xb,
                           float* __restrict__ sq, int B, int N) {
  int t = blockIdx.x * blockDim.x + threadIdx.x;
  if (t >= B * N) return;
  int b = t / N, n = t - b * N;
  float acc = 0.f;
  unsigned short* dst = xb + (size_t)t * 64;
  for (int o = 0; o < 64; ++o) {
    float f = dec_f32(enc[((size_t)(b * 64 + o)) * N + n]);
    dst[o] = f2bf(f);
    acc += f * f;
  }
  sq[t] = acc;
}

// ===========================================================================
// tail_gemm<PASS>: [128 x 256] x [256 x N] per batch; K gathered from the four
// point-major bf16 maps xb1..xb4 (contiguous b128 fragments).
// ===========================================================================
template <int PASS>
__global__ __launch_bounds__(256) void tail_gemm(
    const unsigned short* __restrict__ Wb,
    const unsigned short* __restrict__ x1, const unsigned short* __restrict__ x2,
    const unsigned short* __restrict__ x3, const unsigned short* __restrict__ x4,
    float* __restrict__ stats, const float* __restrict__ scsh,
    float* __restrict__ out, int B, int N) {
  const int O = 128, Cin = 256;
  const int lane = threadIdx.x & 31;
  const int wid  = (blockIdx.x * blockDim.x + threadIdx.x) >> 5;
  const int coltiles = N >> 4;
  const int perB = coltiles * (O >> 4);
  const int b = wid / perB;
  if (b >= B) return;
  const int rem  = wid - b * perB;
  const int row0 = (rem / coltiles) << 4;
  const int n    = ((rem % coltiles) << 4) + (lane & 15);
  const int m    = lane & 15;
  const int coA  = (lane & 16) ? 16 : 0;
  const int coB  = (lane & 16) ? 16 : 0;
  const unsigned short* wrow = Wb + (size_t)(row0 + m) * Cin;
  const size_t pbase = ((size_t)(b * N) + n) * 64;

  v8f c = {};
#pragma unroll
  for (int kb = 0; kb < Cin; kb += 32) {
    v16bf a = load_afrag(wrow + kb + coA);
    const int cc = kb + coB;                // 16-run start within concat dim
    const unsigned short* src = (cc < 64) ? x1 : (cc < 128) ? x2 : (cc < 192) ? x3 : x4;
    v16bf bb = load_bfrag(src + pbase + (cc & 63));
    c = __builtin_amdgcn_wmma_f32_16x16x32_bf16(false, a, false, bb, (short)0, c, false, false);
  }

  if (PASS == 1) {
#pragma unroll
    for (int i = 0; i < 8; ++i) {
      float vs = c[i], vq = c[i] * c[i];
#pragma unroll
      for (int msk = 1; msk < 16; msk <<= 1) { vs += __shfl_xor(vs, msk); vq += __shfl_xor(vq, msk); }
      if ((lane & 15) == 0) {
        int rowg = row0 + i + ((lane & 16) ? 8 : 0);
        atomicAdd(&stats[rowg], vs);
        atomicAdd(&stats[O + rowg], vq);
      }
    }
  } else {
#pragma unroll
    for (int i = 0; i < 8; ++i) {
      int rowg = row0 + i + ((lane & 16) ? 8 : 0);
      float v = c[i] * scsh[rowg] + scsh[O + rowg];
      v = (v >= 0.0f) ? v : 0.2f * v;
      out[((size_t)(b * O + rowg)) * N + n] = v;
    }
  }
}

// ===========================================================================
// Host orchestration
// ===========================================================================
extern "C" void kernel_launch(void* const* d_in, const int* in_sizes, int n_in,
                              void* d_out, int out_size, void* d_ws, size_t ws_size,
                              hipStream_t stream) {
  (void)in_sizes; (void)n_in; (void)out_size; (void)ws_size;
  const int B = 8, N = 2048, C0 = 6;
  const int NK = N * KNN_K;

  const float* x   = (const float*)d_in[0];
  const float* w0a = (const float*)d_in[1];
  const float* g0a = (const float*)d_in[2];
  const float* b0a = (const float*)d_in[3];
  const float* w0b = (const float*)d_in[4];
  const float* g0b = (const float*)d_in[5];
  const float* b0b = (const float*)d_in[6];
  const float* w1a = (const float*)d_in[7];
  const float* g1a = (const float*)d_in[8];
  const float* b1a = (const float*)d_in[9];
  const float* w1b = (const float*)d_in[10];
  const float* g1b = (const float*)d_in[11];
  const float* b1b = (const float*)d_in[12];
  const float* w2  = (const float*)d_in[13];
  const float* g2  = (const float*)d_in[14];
  const float* b2  = (const float*)d_in[15];
  const float* w3  = (const float*)d_in[16];
  const float* g3  = (const float*)d_in[17];
  const float* b3  = (const float*)d_in[18];
  const float* wt  = (const float*)d_in[19];
  const float* gt  = (const float*)d_in[20];
  const float* bt  = (const float*)d_in[21];
  float* out = (float*)d_out;

  // ---- workspace carve (~58 MB total, L2-resident) -------------------------
  char* p = (char*)d_ws;
  auto carve = [&](size_t bytes) { void* r = (void*)p; p += (bytes + 255) & ~(size_t)255; return r; };
  unsigned short* F0  = (unsigned short*)carve((size_t)B * N * 64 * 2);   // bf16 input feats
  unsigned short* xb1 = (unsigned short*)carve((size_t)B * N * 64 * 2);   // bf16 block outs
  unsigned short* xb2 = (unsigned short*)carve((size_t)B * N * 64 * 2);
  unsigned short* xb3 = (unsigned short*)carve((size_t)B * N * 64 * 2);
  unsigned short* xb4 = (unsigned short*)carve((size_t)B * N * 64 * 2);
  float*          sq  = (float*)carve((size_t)B * N * 4);
  int*            nbr = (int*)carve((size_t)B * N * KNN_K * 4);
  unsigned short* tA  = (unsigned short*)carve((size_t)B * NK * 64 * 2);  // bf16 edge act.
  unsigned int*   enc = (unsigned int*)carve((size_t)B * 64 * N * 4);
  unsigned short* Wb  = (unsigned short*)carve((size_t)128 * 256 * 2);    // bf16 weights
  float*          stats = (float*)carve(2 * 128 * 4);
  float*          scsh  = (float*)carve(2 * 128 * 4);

  const int gemmBlocks = (B * 4 * (NK / 16)) / 8;   // 8 waves/block of 256 thr
  const int tailBlocks = (B * 8 * (N / 16)) / 8;
  const int ptBlocks   = (B * N) / 256;
  const float invcntE  = 1.0f / (float)(B * N * KNN_K);
  const float invcntT  = 1.0f / (float)(B * N);
  const dim3 knnGrid(N / 16, B);

  auto cvtW = [&](const float* W, int O, int Cin, int Keff) {
    cvt_weights<<<(O * Keff + 255) / 256, 256, 0, stream>>>(W, Wb, O, Cin, Keff);
  };

  // MODE-0 block-0 conv layer (Cin=6, Keff=32): stats -> finalize -> apply
  auto conv_m0 = [&](const float* W, const float* g, const float* bb) {
    cvtW(W, 64, 12, 32);
    hipMemsetAsync(stats, 0, 1024, stream);
    edge_gemm<0, 1><<<gemmBlocks, 256, 0, stream>>>(Wb, x, nbr, nullptr, nullptr,
        stats, nullptr, nullptr, nullptr, B, N, 64, 32);
    bn_finalize<<<1, 64, 0, stream>>>(stats, g, bb, scsh, invcntE, 64);
    edge_gemm<0, 2><<<gemmBlocks, 256, 0, stream>>>(Wb, x, nbr, nullptr, nullptr,
        nullptr, scsh, tA, nullptr, B, N, 64, 32);
  };
  // MODE-1 edge-feature conv layer (Cin=64, Keff=128) from bf16 map Fs
  auto conv_m1 = [&](const float* W, const unsigned short* Fs, const float* g,
                     const float* bb, unsigned short* tO, unsigned int* eO) {
    cvtW(W, 64, 128, 128);
    hipMemsetAsync(stats, 0, 1024, stream);
    edge_gemm<1, 1><<<gemmBlocks, 256, 0, stream>>>(Wb, nullptr, nbr, Fs, nullptr,
        stats, nullptr, nullptr, nullptr, B, N, 64, 128);
    bn_finalize<<<1, 64, 0, stream>>>(stats, g, bb, scsh, invcntE, 64);
    if (eO) hipMemsetAsync(eO, 0, (size_t)B * 64 * N * 4, stream);
    edge_gemm<1, 2><<<gemmBlocks, 256, 0, stream>>>(Wb, nullptr, nbr, Fs, nullptr,
        nullptr, scsh, tO, eO, B, N, 64, 128);
  };
  // MODE-2 dense-t conv layer (Keff=64)
  auto conv_m2 = [&](const float* W, const float* g, const float* bb, unsigned int* eO) {
    cvtW(W, 64, 64, 64);
    hipMemsetAsync(stats, 0, 1024, stream);
    edge_gemm<2, 1><<<gemmBlocks, 256, 0, stream>>>(Wb, nullptr, nullptr, nullptr, tA,
        stats, nullptr, nullptr, nullptr, B, N, 64, 64);
    bn_finalize<<<1, 64, 0, stream>>>(stats, g, bb, scsh, invcntE, 64);
    hipMemsetAsync(eO, 0, (size_t)B * 64 * N * 4, stream);
    edge_gemm<2, 2><<<gemmBlocks, 256, 0, stream>>>(Wb, nullptr, nullptr, nullptr, tA,
        nullptr, scsh, nullptr, eO, B, N, 64, 64);
  };

  // ---- block 0: knn(x) -> conv(12->64) -> conv(64->64) -> max -> xb1 -------
  prep_input<<<ptBlocks, 256, 0, stream>>>(x, F0, sq, B, N, C0);
  knn_kernel<<<knnGrid, 256, 0, stream>>>(F0, sq, nbr, N);
  conv_m0(w0a, g0a, b0a);
  conv_m2(w0b, g0b, b0b, enc);
  decode_max<<<ptBlocks, 256, 0, stream>>>(enc, xb1, sq, B, N);

  // ---- block 1: knn(xb1) -> conv(128->64) -> conv(64->64) -> max -> xb2 ----
  knn_kernel<<<knnGrid, 256, 0, stream>>>(xb1, sq, nbr, N);
  conv_m1(w1a, xb1, g1a, b1a, tA, nullptr);
  conv_m2(w1b, g1b, b1b, enc);
  decode_max<<<ptBlocks, 256, 0, stream>>>(enc, xb2, sq, B, N);

  // ---- block 2: knn(xb2) -> conv(128->64) -> max -> xb3 --------------------
  knn_kernel<<<knnGrid, 256, 0, stream>>>(xb2, sq, nbr, N);
  conv_m1(w2, xb2, g2, b2, nullptr, enc);
  decode_max<<<ptBlocks, 256, 0, stream>>>(enc, xb3, sq, B, N);

  // ---- block 3: knn(xb3) -> conv(128->64) -> max -> xb4 --------------------
  knn_kernel<<<knnGrid, 256, 0, stream>>>(xb3, sq, nbr, N);
  conv_m1(w3, xb3, g3, b3, nullptr, enc);
  decode_max<<<ptBlocks, 256, 0, stream>>>(enc, xb4, sq, B, N);

  // ---- tail: concat -> 256->128 conv + BN + LeakyReLU ----------------------
  cvtW(wt, 128, 256, 256);
  hipMemsetAsync(stats, 0, 1024, stream);
  tail_gemm<1><<<tailBlocks, 256, 0, stream>>>(Wb, xb1, xb2, xb3, xb4, stats, nullptr,
                                               nullptr, B, N);
  bn_finalize<<<1, 128, 0, stream>>>(stats, gt, bt, scsh, invcntT, 128);
  tail_gemm<2><<<tailBlocks, 256, 0, stream>>>(Wb, xb1, xb2, xb3, xb4, nullptr, scsh,
                                               out, B, N);
}
